// Generator_55207509623221
// MI455X (gfx1250) — compile-verified, matrix operand order checked by reading, not run on hardware
//
#include <hip/hip_runtime.h>

// ---------------- problem constants (match reference) ----------------
constexpr int NB    = 32;
constexpr int NH    = 14;
constexpr int NW    = 14;
constexpr int NP    = NH * NW;        // 196
constexpr int NENC  = 2048;
constexpr int NT    = 21;
constexpr int NSTEP = NT - 1;         // 20
constexpr int NV    = 10000;
constexpr int NE    = 512;
constexpr int NA    = 512;
constexpr int NL    = 512;
constexpr int NX    = NE + NENC;      // 2560
constexpr int NXH   = NX + NL;        // 3072 : [emb | awe*beta | h] concat
constexpr int NZ    = 4 * NL;         // 2048

typedef __bf16 bf16;
typedef __attribute__((ext_vector_type(8)))  __bf16 v8bf;
typedef __attribute__((ext_vector_type(16))) __bf16 v16bf;
typedef __attribute__((ext_vector_type(8)))  float  v8f;

#define CAT16(lo, hi) __builtin_shufflevector(lo, hi, 0,1,2,3,4,5,6,7,8,9,10,11,12,13,14,15)

// ---------------------------------------------------------------------
// bf16 WMMA GEMM, software-pipelined (double-buffered) K loop.
// C[M,N](f32) = A[M,K](bf16 row-major) x B, with B pre-transposed as
// Bt[N,K](bf16 row-major).  Requires K % 64 == 0.  Optional bias[N] and
// per-row mask.  One wave = one 16x16 tile via v_wmma_f32_16x16x32_bf16.
// ---------------------------------------------------------------------
__global__ __launch_bounds__(256)
void gemm_bf16_wmma(const bf16* __restrict__ A, const bf16* __restrict__ Bt,
                    float* __restrict__ C, long ldc,
                    const float* __restrict__ bias,
                    const float* __restrict__ rowmask,
                    int M, int N, int K)
{
    const int lane  = threadIdx.x & 31;
    const int wave  = threadIdx.x >> 5;
    const int wpb   = blockDim.x >> 5;
    const int mtile = blockIdx.y * wpb + wave;
    if ((mtile << 4) >= M) return;                 // wave-uniform exit
    const int row0 = mtile << 4;
    const int col0 = blockIdx.x << 4;

    const int rr = lane & 15;                      // M (A) / N (B) within tile
    const int kb = (lane >> 4) << 3;               // 0 or 8 (16-bit WMMA layout)
    const bf16* Ap = A  + (long)(row0 + rr) * K + kb;
    const bf16* Bp = Bt + (long)(col0 + rr) * K + kb;

    const int ccol  = lane & 15;
    const int rbase = (lane >> 4) << 3;

    v8f acc = {};

    // stage-0 preload
    v8bf a0lo = *reinterpret_cast<const v8bf*>(Ap);
    v8bf a0hi = *reinterpret_cast<const v8bf*>(Ap + 16);
    v8bf b0lo = *reinterpret_cast<const v8bf*>(Bp);
    v8bf b0hi = *reinterpret_cast<const v8bf*>(Bp + 16);

    for (int k0 = 0; k0 < K; k0 += 64) {
        __builtin_prefetch(Ap + k0 + 128, 0, 0);
        __builtin_prefetch(Bp + k0 + 128, 0, 0);
        // stage-1 loads (k0+32) issued before consuming stage-0
        v8bf a1lo = *reinterpret_cast<const v8bf*>(Ap + k0 + 32);
        v8bf a1hi = *reinterpret_cast<const v8bf*>(Ap + k0 + 48);
        v8bf b1lo = *reinterpret_cast<const v8bf*>(Bp + k0 + 32);
        v8bf b1hi = *reinterpret_cast<const v8bf*>(Bp + k0 + 48);

        {
            v16bf a = CAT16(a0lo, a0hi);
            v16bf b = CAT16(b0lo, b0hi);
            acc = __builtin_amdgcn_wmma_f32_16x16x32_bf16(
                      false, a, false, b, (short)0, acc, false, false);
        }
        if (k0 + 64 < K) {                         // stage-0 loads for next iter
            a0lo = *reinterpret_cast<const v8bf*>(Ap + k0 + 64);
            a0hi = *reinterpret_cast<const v8bf*>(Ap + k0 + 80);
            b0lo = *reinterpret_cast<const v8bf*>(Bp + k0 + 64);
            b0hi = *reinterpret_cast<const v8bf*>(Bp + k0 + 80);
        }
        {
            v16bf a = CAT16(a1lo, a1hi);
            v16bf b = CAT16(b1lo, b1hi);
            acc = __builtin_amdgcn_wmma_f32_16x16x32_bf16(
                      false, a, false, b, (short)0, acc, false, false);
        }
    }

#pragma unroll
    for (int j = 0; j < 8; ++j) {
        float v = acc[j];
        if (bias)    v += bias[col0 + ccol];
        if (rowmask) v *= rowmask[row0 + rbase + j];
        C[(long)(row0 + rbase + j) * ldc + col0 + ccol] = v;
    }
}

// ---------------- helpers: conversions / transposes ----------------
__global__ void f32_to_bf16_k(const float* __restrict__ in, bf16* __restrict__ out, long n)
{
    long i = (long)blockIdx.x * blockDim.x + threadIdx.x;
    if (i < n) out[i] = (bf16)in[i];
}

// W[K,N] f32 (row-major)  ->  Wt[N,K] bf16 (row-major) == B transposed
__global__ void xpose_bf16_k(const float* __restrict__ W, bf16* __restrict__ Wt,
                             int K, int N)
{
    long i = (long)blockIdx.x * blockDim.x + threadIdx.x;
    if (i >= (long)N * K) return;
    int n = (int)(i / K), k = (int)(i % K);
    Wt[i] = (bf16)W[(long)k * N + n];
}

// Concatenated transpose: WluT[NZ, NXH] <- [Wl(NX,NZ) ; Ul(NL,NZ)]^T
__global__ void xpose_wlu_k(const float* __restrict__ Wl, const float* __restrict__ Ul,
                            bf16* __restrict__ WluT)
{
    long i = (long)blockIdx.x * blockDim.x + threadIdx.x;
    if (i >= (long)NZ * NXH) return;
    int n = (int)(i / NXH), k = (int)(i % NXH);
    float v = (k < NX) ? Wl[(long)k * NZ + n] : Ul[(long)(k - NX) * NZ + n];
    WluT[i] = (bf16)v;
}

// write h (f32 [B,NL]) into tail slot of concat x buffer, bf16
__global__ void h_to_x_k(const float* __restrict__ h, bf16* __restrict__ xcat)
{
    int b = blockIdx.x;
    int u = blockIdx.y * blockDim.x + threadIdx.x;
    xcat[(long)b * NXH + NX + u] = (bf16)h[(long)b * NL + u];
}

// ---------------- ordering (packed-sequence sort) ----------------
__global__ void order_k(const int* __restrict__ seqs, int* __restrict__ order,
                        int* __restrict__ lens, float* __restrict__ maskmat)
{
    __shared__ int len[NB];
    int b = threadIdx.x;
    if (b < NB) {
        int c = 0;
        for (int t = 0; t < NT; ++t) c += (seqs[b * NT + t] != 0) ? 1 : 0;
        len[b] = c;
    }
    __syncthreads();
    if (b < NB) {
        int pos = 0;                                  // stable descending rank
        for (int j = 0; j < NB; ++j)
            if (len[j] > len[b] || (len[j] == len[b] && j < b)) pos++;
        order[pos] = b;
        lens[pos]  = len[b];
    }
    __syncthreads();
    if (b < NB) {
        int Lb = lens[b] - 1;
        for (int s = 0; s < NSTEP; ++s)
            maskmat[s * NB + b] = (Lb > s) ? 1.0f : 0.0f;
    }
}

__global__ void write_order_k(const int* __restrict__ order, float* __restrict__ out)
{
    int b = threadIdx.x;
    if (b < NB) out[b] = (float)order[b];
}

// ---------------- encoder gather + bf16 convert + mean pool ----------------
__global__ void enc_prep_k(const float* __restrict__ enc_in, const int* __restrict__ order,
                           bf16* __restrict__ enc_bf, float* __restrict__ mean,
                           bf16* __restrict__ mean_bf)
{
    int b  = blockIdx.x;
    int ch = blockIdx.y * blockDim.x + threadIdx.x;
    const float* src = enc_in + (long)order[b] * NP * NENC;
    float s = 0.f;
    for (int p = 0; p < NP; ++p) {
        float v = src[(long)p * NENC + ch];
        s += v;
        enc_bf[((long)b * NP + p) * NENC + ch] = (bf16)v;
    }
    float m = s * (1.0f / NP);
    mean[(long)b * NENC + ch]    = m;
    mean_bf[(long)b * NENC + ch] = (bf16)m;
}

// ---------------- attention: e[b,p] = relu(a1+a2) . Wfa + bfa ----------------
__global__ void attn_e_k(const bf16* __restrict__ a1bf, const float* __restrict__ a2,
                         const float* __restrict__ Wfa, const float* __restrict__ bfa,
                         float* __restrict__ e)
{
    long bp = blockIdx.x;
    int  b  = (int)(bp / NP);
    const bf16*  row = a1bf + bp * NA;
    const float* a2r = a2 + (long)b * NA;
    float s = 0.f;
    for (int i = threadIdx.x; i < NA; i += blockDim.x) {
        float v = (float)row[i] + a2r[i];
        v = v > 0.f ? v : 0.f;
        s += v * Wfa[i];
    }
    __shared__ float red[256];
    red[threadIdx.x] = s; __syncthreads();
    for (int off = 128; off > 0; off >>= 1) {
        if (threadIdx.x < off) red[threadIdx.x] += red[threadIdx.x + off];
        __syncthreads();
    }
    if (threadIdx.x == 0) e[bp] = red[0] + bfa[0];
}

// ---------------- softmax over P, also emits masked alphas output ----------------
__global__ void softmax_k(const float* __restrict__ e, float* __restrict__ alpha,
                          float* __restrict__ alphas_out,
                          const float* __restrict__ maskrow, int step)
{
    int b = blockIdx.x, t = threadIdx.x;
    float v = (t < NP) ? e[(long)b * NP + t] : -1e30f;
    __shared__ float red[256];
    red[t] = v; __syncthreads();
    for (int off = 128; off > 0; off >>= 1) {
        if (t < off) red[t] = fmaxf(red[t], red[t + off]);
        __syncthreads();
    }
    float mx = red[0]; __syncthreads();
    float ex = (t < NP) ? __expf(v - mx) : 0.f;
    red[t] = ex; __syncthreads();
    for (int off = 128; off > 0; off >>= 1) {
        if (t < off) red[t] += red[t + off];
        __syncthreads();
    }
    float inv = 1.0f / red[0];
    if (t < NP) {
        float al = ex * inv;
        alpha[(long)b * NP + t] = al;
        alphas_out[((long)b * NSTEP + step) * NP + t] = al * maskrow[b];
    }
}

// ---------------- awe[b,c] = sum_p alpha[b,p] * enc[b,p,c] ----------------
__global__ void awe_k(const bf16* __restrict__ enc_bf, const float* __restrict__ alpha,
                      float* __restrict__ awe)
{
    int b  = blockIdx.x;
    int ch = blockIdx.y * blockDim.x + threadIdx.x;
    __shared__ float al[NP];
    for (int p = threadIdx.x; p < NP; p += blockDim.x) al[p] = alpha[(long)b * NP + p];
    __syncthreads();
    const bf16* basep = enc_bf + (long)b * NP * NENC + ch;
    float s = 0.f;
    for (int p = 0; p < NP; ++p) s += al[p] * (float)basep[(long)p * NENC];
    awe[(long)b * NENC + ch] = s;
}

// ---------------- x[:, :NX] = [emb[tok], awe * sigmoid(betaPre)] ----------------
__global__ void build_x_k(const float* __restrict__ emb, const int* __restrict__ seqs,
                          const int* __restrict__ order, const float* __restrict__ awe,
                          const float* __restrict__ betaPre, bf16* __restrict__ xcat, int step)
{
    int b = blockIdx.x;
    int tok = seqs[(long)order[b] * NT + step];
    const float* er = emb + (long)tok * NE;
    bf16* xr = xcat + (long)b * NXH;
    for (int i = threadIdx.x; i < NE; i += blockDim.x) xr[i] = (bf16)er[i];
    for (int i = threadIdx.x; i < NENC; i += blockDim.x) {
        float beta = 1.0f / (1.0f + __expf(-betaPre[(long)b * NENC + i]));
        xr[NE + i] = (bf16)(awe[(long)b * NENC + i] * beta);
    }
}

// ---------------- LSTM gate update; writes c (in-place), cbf, h->xcat ----------------
__global__ void lstm_k(const float* __restrict__ z, float* __restrict__ c,
                       float* __restrict__ h_out, bf16* __restrict__ c_bf,
                       bf16* __restrict__ xcat, const float* __restrict__ maskrow)
{
    int b = blockIdx.x;
    int u = blockIdx.y * blockDim.x + threadIdx.x;
    const float* zr = z + (long)b * NZ;
    float zi = zr[u], zf = zr[NL + u], zg = zr[2 * NL + u], zo = zr[3 * NL + u];
    float si = 1.0f / (1.0f + __expf(-zi));
    float sf = 1.0f / (1.0f + __expf(-zf));
    float so = 1.0f / (1.0f + __expf(-zo));
    float cn = sf * c[(long)b * NL + u] + si * tanhf(zg);
    float hn = so * tanhf(cn);
    float mf = maskrow[b];
    cn *= mf; hn *= mf;
    c[(long)b * NL + u]     = cn;                  // in-place (one elem per thread)
    h_out[(long)b * NL + u] = hn;
    c_bf[(long)b * NL + u]  = (bf16)cn;
    xcat[(long)b * NXH + NX + u] = (bf16)hn;       // h feeds next step's fused z GEMM
}

// =====================================================================
extern "C" void kernel_launch(void* const* d_in, const int* in_sizes, int n_in,
                              void* d_out, int out_size, void* d_ws, size_t ws_size,
                              hipStream_t stream)
{
    const float* enc_in = (const float*)d_in[0];
    const int*   seqs   = (const int*)  d_in[1];
    const float* emb    = (const float*)d_in[2];
    const float* Wea = (const float*)d_in[3];  const float* bea = (const float*)d_in[4];
    const float* Wga = (const float*)d_in[5];  const float* bga = (const float*)d_in[6];
    const float* Wfa = (const float*)d_in[7];  const float* bfa = (const float*)d_in[8];
    const float* Wl  = (const float*)d_in[9];  const float* Ul  = (const float*)d_in[10];
    const float* bl  = (const float*)d_in[11];
    const float* Wim = (const float*)d_in[12]; const float* bim = (const float*)d_in[13];
    const float* Wic = (const float*)d_in[14]; const float* bic = (const float*)d_in[15];
    const float* Wb  = (const float*)d_in[16]; const float* bb  = (const float*)d_in[17];
    const float* Wo  = (const float*)d_in[18]; const float* bo  = (const float*)d_in[19];

    // ---- output layout: preds [B,STEP,V] | alphas [B,STEP,P] | order [B] ----
    float* preds_out  = (float*)d_out;
    float* alphas_out = preds_out  + (size_t)NB * NSTEP * NV;
    float* order_out  = alphas_out + (size_t)NB * NSTEP * NP;

    // ---- workspace carve-up (256B aligned) ----
    char* wbase = (char*)d_ws; size_t woff = 0;
    auto alloc = [&](size_t bytes) -> void* {
        void* p = wbase + woff;
        woff = (woff + bytes + 255) & ~(size_t)255;
        return p;
    };
    bf16* WeaT  = (bf16*)alloc((size_t)NA   * NENC * 2);
    bf16* WgaT  = (bf16*)alloc((size_t)NA   * NL   * 2);
    bf16* WluT  = (bf16*)alloc((size_t)NZ   * NXH  * 2);
    bf16* WimT  = (bf16*)alloc((size_t)NL   * NENC * 2);
    bf16* WicT  = (bf16*)alloc((size_t)NL   * NENC * 2);
    bf16* WbT   = (bf16*)alloc((size_t)NENC * NL   * 2);
    bf16* WoT   = (bf16*)alloc((size_t)NV   * NL   * 2);
    bf16* encbf = (bf16*)alloc((size_t)NB * NP * NENC * 2);
    float* a1    = (float*)alloc((size_t)NB * NP * NA * 4);
    bf16*  a1bf  = (bf16*) alloc((size_t)NB * NP * NA * 2);
    float* mean  = (float*)alloc((size_t)NB * NENC * 4);
    bf16*  meanbf= (bf16*) alloc((size_t)NB * NENC * 2);
    float* htmp  = (float*)alloc((size_t)NB * NL * 4);
    float* c32   = (float*)alloc((size_t)NB * NL * 4);
    bf16*  cbf   = (bf16*) alloc((size_t)NB * NL * 2);
    float* a2    = (float*)alloc((size_t)NB * NA * 4);
    float* evec  = (float*)alloc((size_t)NB * NP * 4);
    float* alpha = (float*)alloc((size_t)NB * NP * 4);
    float* awe   = (float*)alloc((size_t)NB * NENC * 4);
    float* betaP = (float*)alloc((size_t)NB * NENC * 4);
    bf16*  xcat  = (bf16*) alloc((size_t)NB * NXH * 2);
    float* z     = (float*)alloc((size_t)NB * NZ * 4);
    int*   order = (int*)  alloc((size_t)NB * 4);
    int*   lens  = (int*)  alloc((size_t)NB * 4);
    float* maskm = (float*)alloc((size_t)NSTEP * NB * 4);
    (void)ws_size; (void)in_sizes; (void)n_in; (void)out_size;

    auto gemm = [&](const bf16* Aop, const bf16* Bt, float* Cout, long ldc,
                    const float* bias, const float* rmask,
                    int M, int N, int K) {
        int mtiles = M / 16;
        int wpb = mtiles < 8 ? mtiles : 8;
        dim3 grid(N / 16, (mtiles + wpb - 1) / wpb);
        gemm_bf16_wmma<<<grid, wpb * 32, 0, stream>>>(Aop, Bt, Cout, ldc,
                                                      bias, rmask, M, N, K);
    };
    auto xpose = [&](const float* W, bf16* Wt, int K, int N) {
        long n = (long)K * N;
        xpose_bf16_k<<<(unsigned)((n + 255) / 256), 256, 0, stream>>>(W, Wt, K, N);
    };

    // ---------------- init phase ----------------
    order_k<<<1, 32, 0, stream>>>(seqs, order, lens, maskm);
    write_order_k<<<1, 32, 0, stream>>>(order, order_out);

    xpose(Wea, WeaT, NENC, NA);
    xpose(Wga, WgaT, NL,   NA);
    xpose(Wim, WimT, NENC, NL);
    xpose(Wic, WicT, NENC, NL);
    xpose(Wb,  WbT,  NL,   NENC);
    xpose(Wo,  WoT,  NL,   NV);
    {
        long n = (long)NZ * NXH;
        xpose_wlu_k<<<(unsigned)((n + 255) / 256), 256, 0, stream>>>(Wl, Ul, WluT);
    }

    enc_prep_k<<<dim3(NB, NENC / 256), 256, 0, stream>>>(enc_in, order, encbf, mean, meanbf);

    // h0 / c0 from mean-pooled features
    gemm(meanbf, WimT, htmp, NL, bim, nullptr, NB, NL, NENC);
    gemm(meanbf, WicT, c32,  NL, bic, nullptr, NB, NL, NENC);
    h_to_x_k<<<dim3(NB, NL / 256), 256, 0, stream>>>(htmp, xcat);
    f32_to_bf16_k<<<(NB * NL + 255) / 256, 256, 0, stream>>>(c32, cbf, (long)NB * NL);

    // a1 = enc @ Wea + bea  (computed once), then cached as bf16 for step loop
    gemm(encbf, WeaT, a1, NA, bea, nullptr, NB * NP, NA, NENC);
    f32_to_bf16_k<<<(unsigned)(((long)NB * NP * NA + 255) / 256), 256, 0, stream>>>(
        a1, a1bf, (long)NB * NP * NA);

    // ---------------- decode loop ----------------
    for (int step = 0; step < NSTEP; ++step) {
        const float* mrow = maskm + (size_t)step * NB;

        gemm(cbf, WgaT, a2, NA, bga, nullptr, NB, NA, NL);
        attn_e_k<<<NB * NP, 256, 0, stream>>>(a1bf, a2, Wfa, bfa, evec);
        softmax_k<<<NB, 256, 0, stream>>>(evec, alpha, alphas_out, mrow, step);
        awe_k<<<dim3(NB, NENC / 256), 256, 0, stream>>>(encbf, alpha, awe);

        gemm(cbf, WbT, betaP, NENC, bb, nullptr, NB, NENC, NL);
        build_x_k<<<NB, 256, 0, stream>>>(emb, seqs, order, awe, betaP, xcat, step);

        // fused z = [x|h] @ [Wl;Ul]^T + bl   (K = 3072)
        gemm(xcat, WluT, z, NZ, bl, nullptr, NB, NZ, NXH);

        lstm_k<<<dim3(NB, NL / 256), 256, 0, stream>>>(z, c32, htmp, cbf, xcat, mrow);

        // pred = (c_new @ Wo + bo) * mask  -> strided rows into preds_out
        gemm(cbf, WoT, preds_out + (size_t)step * NV, (long)NSTEP * NV,
             bo, mrow, NB, NV, NL);
    }
}
